// MVGFormerHand_13804024889502
// MI455X (gfx1250) — compile-verified, matrix operand order checked by reading, not run on hardware
//
#include <hip/hip_runtime.h>

typedef __attribute__((ext_vector_type(16))) __bf16 v16bf;
typedef __attribute__((ext_vector_type(8)))  float  v8f;
typedef __attribute__((ext_vector_type(4)))  unsigned int uint32x4;
typedef __attribute__((ext_vector_type(8)))  int int32x8;
typedef __attribute__((ext_vector_type(4)))  int int32x4;

union V16BF { v16bf v; unsigned int u[8]; };

__device__ __forceinline__ unsigned short bf16_bits(float x) {
    unsigned int u = __float_as_uint(x);
    unsigned int r = u + 0x7FFFu + ((u >> 16) & 1u);
    return (unsigned short)(r >> 16);
}
__device__ __forceinline__ unsigned int pack_bf16(float lo, float hi) {
    return (unsigned int)bf16_bits(lo) | ((unsigned int)bf16_bits(hi) << 16);
}

// ---------------- elementwise helpers ----------------
__global__ void k_cvt(const float* __restrict__ s, unsigned short* __restrict__ d, int n) {
    int i = blockIdx.x * 256 + threadIdx.x;
    if (i < n) d[i] = bf16_bits(s[i]);
}

__global__ void k_add(const float* __restrict__ a, const float* __restrict__ b,
                      float* __restrict__ c, int n) {
    int i = blockIdx.x * 256 + threadIdx.x;
    if (i < n) c[i] = a[i] + b[i];
}

// ---------------- residual + layernorm (row = 256 ch) ----------------
__global__ __launch_bounds__(256) void k_ln(float* __restrict__ x, const float* __restrict__ y,
                                            const float* __restrict__ g, const float* __restrict__ bb) {
    __shared__ float red[256];
    int row = blockIdx.x, c = threadIdx.x;
    size_t idx = (size_t)row * 256 + c;
    float val = x[idx] + y[idx];
    red[c] = val; __syncthreads();
    for (int s = 128; s > 0; s >>= 1) { if (c < s) red[c] += red[c + s]; __syncthreads(); }
    float mean = red[0] * (1.f / 256.f);
    __syncthreads();
    float d = val - mean;
    red[c] = d * d; __syncthreads();
    for (int s = 128; s > 0; s >>= 1) { if (c < s) red[c] += red[c + s]; __syncthreads(); }
    float var = red[0] * (1.f / 256.f);
    x[idx] = d * rsqrtf(var + 1e-5f) * g[c] + bb[c];
}

// ---------------- WMMA GEMM with TDM-staged A tile ----------------
// C[M,N] = A[M,K] * W[N,K]^T + bias.  M mult of 16, N mult of 64, K mult of 32 (K<=1024).
// One 4-wave workgroup per 16(M) x 64(N) tile.  Wave 0 DMAs the 16xK A panel into LDS
// via the Tensor Data Mover; all 4 waves consume it and stream their own W panel.
__global__ __launch_bounds__(128) void k_gemm(const unsigned short* __restrict__ A,
                                              const unsigned short* __restrict__ W,
                                              const float* __restrict__ bias,
                                              float* __restrict__ C,
                                              int K, int ldc, int relu) {
    __shared__ unsigned short sA[16 * 1024];   // 32 KB: 16 rows x K (K<=1024) bf16
    const int tid  = threadIdx.x;
    const int lane = tid & 31;
    const int wave = tid >> 5;
    const int l15  = lane & 15;
    const int half = lane >> 4;
    const int rowbase = blockIdx.y * 16;

    if (wave == 0) {
        // Build D# descriptor (CDNA5 ISA ch.8): 2D tile, 16 rows x K elems, 2B elems.
        unsigned long long gaddr =
            (unsigned long long)(const void*)(A + (size_t)rowbase * K);
        uint32x4 g0;
        g0[0] = 1u;                                               // count=1, no gather
        g0[1] = 0u;                                               // lds_addr = 0 (sA base)
        g0[2] = (unsigned int)(gaddr & 0xFFFFFFFFu);              // global_addr[31:0]
        g0[3] = (unsigned int)((gaddr >> 32) & 0x1FFFFFFu)        // global_addr[56:32]
              | (2u << 30);                                       // type = 2 (image)
        int32x8 g1;
        g1[0] = 0x00010000;                                       // data_size = 2 bytes
        g1[1] = (int)((unsigned)(K & 0xFFFF) << 16);              // tensor_dim0[15:0]
        g1[2] = (int)(((unsigned)K >> 16) | (16u << 16));         // dim0 hi | tensor_dim1=16
        g1[3] = (int)((unsigned)(K & 0xFFFF) << 16);              // tile_dim0 = K
        g1[4] = 16;                                               // tile_dim1 = 16
        g1[5] = K;                                                // tensor_dim0_stride lo
        g1[6] = (int)((unsigned)(K & 0xFFFF) << 16);              // dim1_stride lo (unused 2D)
        g1[7] = 0;
        int32x4 z4 = {0, 0, 0, 0};
        int32x8 z8 = {0, 0, 0, 0, 0, 0, 0, 0};
        __builtin_amdgcn_tensor_load_to_lds(g0, g1, z4, z4, z8, 0);
        __builtin_amdgcn_s_wait_tensorcnt(0);
    }
    // The TDM write to LDS is invisible to alias analysis: force the compiler to
    // treat sA as (potentially) written before any lane reads it.
    {
        unsigned short* sap = &sA[0];
        asm volatile("" : "+r"(sap) : : "memory");
    }
    __syncthreads();

    const int col = blockIdx.x * 64 + wave * 16 + l15;   // W row / output column
    v8f acc = {0.f, 0.f, 0.f, 0.f, 0.f, 0.f, 0.f, 0.f};
    for (int kb = 0; kb < K; kb += 32) {
        if (kb + 32 < K) __builtin_prefetch(W + (size_t)col * K + kb + 32, 0, 0);
        V16BF a, b;
#pragma unroll
        for (int i = 0; i < 8; ++i) {
            int k0 = kb + (i >> 2) * 16 + half * 8 + (i & 3) * 2;
            a.u[i] = *(const unsigned int*)(sA + l15 * K + k0);
            b.u[i] = *(const unsigned int*)(W + (size_t)col * K + k0);
        }
        acc = __builtin_amdgcn_wmma_f32_16x16x32_bf16(false, a.v, false, b.v,
                                                      (short)0, acc, false, false);
    }
    const int m0 = rowbase + half * 8;
#pragma unroll
    for (int r = 0; r < 8; ++r) {
        float v = acc[r] + (bias ? bias[col] : 0.f);
        if (relu) v = fmaxf(v, 0.f);
        C[(size_t)(m0 + r) * ldc + col] = v;
    }
}

// ---------------- WMMA attention: one wave per (b, head, 16-query tile) ----------------
#define NQ_C 336
#define NQ_PAD 352
__global__ __launch_bounds__(32) void k_attn(const float* __restrict__ qkv, float* __restrict__ out) {
    __shared__ float sc[16][NQ_PAD];
    const int lane = threadIdx.x;
    const int l15  = lane & 15;
    const int half = lane >> 4;
    const int qt = blockIdx.x;  // 0..20
    const int h  = blockIdx.y;  // 0..7
    const int b  = blockIdx.z;  // 0..1
    const float scale = 0.17677669529f;  // 1/sqrt(32)

    V16BF a;
    {
        const float* qrow = qkv + ((size_t)(b * NQ_C + qt * 16 + l15)) * 768 + h * 32;
#pragma unroll
        for (int i = 0; i < 8; ++i) {
            int k0 = (i >> 2) * 16 + half * 8 + (i & 3) * 2;
            a.u[i] = pack_bf16(qrow[k0], qrow[k0 + 1]);
        }
    }
    for (int kt = 0; kt < 21; ++kt) {
        V16BF bm;
        const float* krow = qkv + ((size_t)(b * NQ_C + kt * 16 + l15)) * 768 + 256 + h * 32;
#pragma unroll
        for (int i = 0; i < 8; ++i) {
            int k0 = (i >> 2) * 16 + half * 8 + (i & 3) * 2;
            bm.u[i] = pack_bf16(krow[k0], krow[k0 + 1]);
        }
        v8f c = {0.f, 0.f, 0.f, 0.f, 0.f, 0.f, 0.f, 0.f};
        c = __builtin_amdgcn_wmma_f32_16x16x32_bf16(false, a.v, false, bm.v,
                                                    (short)0, c, false, false);
#pragma unroll
        for (int r = 0; r < 8; ++r) sc[half * 8 + r][kt * 16 + l15] = c[r] * scale;
    }
#pragma unroll
    for (int r = 0; r < 8; ++r) sc[half * 8 + r][NQ_C + l15] = 0.f;
    __syncthreads();

    for (int r = 0; r < 16; ++r) {
        float mx = -3.4e38f;
        for (int i = lane; i < NQ_C; i += 32) mx = fmaxf(mx, sc[r][i]);
#pragma unroll
        for (int o = 16; o > 0; o >>= 1) mx = fmaxf(mx, __shfl_xor(mx, o, 32));
        float sum = 0.f;
        for (int i = lane; i < NQ_C; i += 32) { float e = __expf(sc[r][i] - mx); sc[r][i] = e; sum += e; }
#pragma unroll
        for (int o = 16; o > 0; o >>= 1) sum += __shfl_xor(sum, o, 32);
        float inv = 1.f / sum;
        for (int i = lane; i < NQ_C; i += 32) sc[r][i] *= inv;
    }
    __syncthreads();

    for (int nh = 0; nh < 2; ++nh) {
        v8f c = {0.f, 0.f, 0.f, 0.f, 0.f, 0.f, 0.f, 0.f};
        for (int ch = 0; ch < 11; ++ch) {
            V16BF ap, bp;
#pragma unroll
            for (int i = 0; i < 8; ++i) {
                int k0 = ch * 32 + (i >> 2) * 16 + half * 8 + (i & 3) * 2;
                ap.u[i] = pack_bf16(sc[l15][k0], sc[l15][k0 + 1]);
                float v0 = (k0 < NQ_C)
                    ? qkv[((size_t)(b * NQ_C + k0)) * 768 + 512 + h * 32 + nh * 16 + l15] : 0.f;
                float v1 = (k0 + 1 < NQ_C)
                    ? qkv[((size_t)(b * NQ_C + k0 + 1)) * 768 + 512 + h * 32 + nh * 16 + l15] : 0.f;
                bp.u[i] = pack_bf16(v0, v1);
            }
            c = __builtin_amdgcn_wmma_f32_16x16x32_bf16(false, ap.v, false, bp.v,
                                                        (short)0, c, false, false);
        }
#pragma unroll
        for (int r = 0; r < 8; ++r)
            out[((size_t)(b * NQ_C + qt * 16 + half * 8 + r)) * 256 + h * 32 + nh * 16 + l15] = c[r];
    }
}

// ---------------- projection + multilevel bilinear sample + mask fuse ----------------
__global__ __launch_bounds__(256) void k_proj_sample(const float* __restrict__ ref,
                                                     const float* __restrict__ R,
                                                     const float* __restrict__ T,
                                                     const float* __restrict__ Kc,
                                                     const float* __restrict__ f0,
                                                     const float* __restrict__ f1,
                                                     const float* __restrict__ f2,
                                                     float* __restrict__ fused) {
    const int NV = 8;
    __shared__ float sw[8][3][4];
    __shared__ int   so[8][3][4];
    __shared__ float smask[8];
    int b = blockIdx.y, q = blockIdx.x, c = threadIdx.x;
    if (threadIdx.x < NV * 3) {
        int v = threadIdx.x / 3, lvl = threadIdx.x % 3;
        const float* p  = ref + ((size_t)b * 336 + q) * 3;
        const float* Rv = R + ((size_t)(b * NV + v)) * 9;
        const float* Tv = T + ((size_t)(b * NV + v)) * 3;
        const float* Kv = Kc + ((size_t)(b * NV + v)) * 9;
        float p0 = Rv[0] * p[0] + Rv[1] * p[1] + Rv[2] * p[2] + Tv[0];
        float p1 = Rv[3] * p[0] + Rv[4] * p[1] + Rv[5] * p[2] + Tv[1];
        float p2 = Rv[6] * p[0] + Rv[7] * p[1] + Rv[8] * p[2] + Tv[2];
        float z = fmaxf(p2, 0.1f);
        float u  = p0 * Kv[0] / z + Kv[2];
        float vv = p1 * Kv[4] / z + Kv[5];
        float un = 2.f * u  / 511.f - 1.f;
        float vn = 2.f * vv / 511.f - 1.f;
        float mk = (un > -1.f && un < 1.f && vn > -1.f && vn < 1.f && p2 > 0.f) ? 1.f : 0.f;
        if (lvl == 0) smask[v] = mk;
        int W = (lvl == 0) ? 64 : ((lvl == 1) ? 32 : 16);
        float x = (un + 1.f) * W * 0.5f - 0.5f;
        float y = (vn + 1.f) * W * 0.5f - 0.5f;
        float x0f = floorf(x), y0f = floorf(y);
        int ix0 = (int)x0f, iy0 = (int)y0f, ix1 = ix0 + 1, iy1 = iy0 + 1;
        float wx1 = x - x0f, wx0 = 1.f - wx1, wy1 = y - y0f, wy0 = 1.f - wy1;
        float v00 = (ix0 >= 0 && ix0 < W && iy0 >= 0 && iy0 < W) ? 1.f : 0.f;
        float v01 = (ix1 >= 0 && ix1 < W && iy0 >= 0 && iy0 < W) ? 1.f : 0.f;
        float v10 = (ix0 >= 0 && ix0 < W && iy1 >= 0 && iy1 < W) ? 1.f : 0.f;
        float v11 = (ix1 >= 0 && ix1 < W && iy1 >= 0 && iy1 < W) ? 1.f : 0.f;
        int ix0c = min(max(ix0, 0), W - 1), ix1c = min(max(ix1, 0), W - 1);
        int iy0c = min(max(iy0, 0), W - 1), iy1c = min(max(iy1, 0), W - 1);
        sw[v][lvl][0] = wy0 * wx0 * v00;  so[v][lvl][0] = iy0c * W + ix0c;
        sw[v][lvl][1] = wy0 * wx1 * v01;  so[v][lvl][1] = iy0c * W + ix1c;
        sw[v][lvl][2] = wy1 * wx0 * v10;  so[v][lvl][2] = iy1c * W + ix0c;
        sw[v][lvl][3] = wy1 * wx1 * v11;  so[v][lvl][3] = iy1c * W + ix1c;
    }
    __syncthreads();
    float cnt = 0.f;
#pragma unroll
    for (int v = 0; v < NV; ++v) cnt += smask[v];
    float acc = 0.f;
    for (int v = 0; v < NV; ++v) {
        if (smask[v] > 0.f) {
            size_t bc = (size_t)(b * NV + v) * 256 + c;
            const float* p0 = f0 + bc * 4096;
            const float* p1 = f1 + bc * 1024;
            const float* p2 = f2 + bc * 256;
            float s = sw[v][0][0] * p0[so[v][0][0]] + sw[v][0][1] * p0[so[v][0][1]]
                    + sw[v][0][2] * p0[so[v][0][2]] + sw[v][0][3] * p0[so[v][0][3]]
                    + sw[v][1][0] * p1[so[v][1][0]] + sw[v][1][1] * p1[so[v][1][1]]
                    + sw[v][1][2] * p1[so[v][1][2]] + sw[v][1][3] * p1[so[v][1][3]]
                    + sw[v][2][0] * p2[so[v][2][0]] + sw[v][2][1] * p2[so[v][2][1]]
                    + sw[v][2][2] * p2[so[v][2][2]] + sw[v][2][3] * p2[so[v][2][3]];
            acc += s * (1.f / 3.f);
        }
    }
    fused[((size_t)b * 336 + q) * 256 + c] = acc / fmaxf(cnt, 1.f);
}

// ---------------- tiny heads ----------------
__global__ void k_pose_head(const float* __restrict__ h2, const float* __restrict__ w,
                            const float* __restrict__ bias, float* __restrict__ ref,
                            float* __restrict__ out, int l) {
    int idx = blockIdx.x * 256 + threadIdx.x;
    if (idx >= 672 * 3) return;
    int row = idx / 3, j = idx % 3;
    const float* hr = h2 + (size_t)row * 256;
    const float* wr = w + (size_t)j * 256;
    float acc = 0.f;
    for (int k = 0; k < 256; ++k) acc += hr[k] * wr[k];
    float nr = ref[idx] + acc + bias[j];
    ref[idx] = nr;
    out[((size_t)l * 672 + row) * 4 + 1 + j] = nr;
}

__global__ void k_cls_head(const float* __restrict__ x, const float* __restrict__ w,
                           const float* __restrict__ bias, float* __restrict__ out, int l) {
    int row = blockIdx.x * 256 + threadIdx.x;
    if (row >= 672) return;
    const float* xr = x + (size_t)row * 256;
    float acc = 0.f;
    for (int k = 0; k < 256; ++k) acc += xr[k] * w[k];
    out[((size_t)l * 672 + row) * 4] = acc + bias[0];
}

// ---------------- host orchestration ----------------
extern "C" void kernel_launch(void* const* d_in, const int* in_sizes, int n_in,
                              void* d_out, int out_size, void* d_ws, size_t ws_size,
                              hipStream_t stream) {
    (void)in_sizes; (void)n_in; (void)out_size; (void)ws_size;
    const float* tgt   = (const float*)d_in[0];
    const float* qpos  = (const float*)d_in[1];
    const float* refp  = (const float*)d_in[2];
    const float* f0    = (const float*)d_in[3];
    const float* f1    = (const float*)d_in[4];
    const float* f2    = (const float*)d_in[5];
    const float* camR  = (const float*)d_in[6];
    const float* camT  = (const float*)d_in[7];
    const float* camK  = (const float*)d_in[8];
    const float* Wqkv  = (const float*)d_in[9];
    const float* bqkv  = (const float*)d_in[10];
    const float* Wo    = (const float*)d_in[11];
    const float* bo    = (const float*)d_in[12];
    const float* ln1g  = (const float*)d_in[13];
    const float* ln1b  = (const float*)d_in[14];
    const float* Wproj = (const float*)d_in[15];
    const float* bproj = (const float*)d_in[16];
    const float* png   = (const float*)d_in[17];
    const float* pnb   = (const float*)d_in[18];
    const float* W1    = (const float*)d_in[19];
    const float* b1    = (const float*)d_in[20];
    const float* W2    = (const float*)d_in[21];
    const float* b2    = (const float*)d_in[22];
    const float* ln2g  = (const float*)d_in[23];
    const float* ln2b  = (const float*)d_in[24];
    const float* pW0   = (const float*)d_in[25];
    const float* pb0   = (const float*)d_in[26];
    const float* pW1   = (const float*)d_in[27];
    const float* pb1   = (const float*)d_in[28];
    const float* pW2   = (const float*)d_in[29];
    const float* pb2   = (const float*)d_in[30];
    const float* clsW  = (const float*)d_in[31];
    const float* clsb  = (const float*)d_in[32];
    float* outp = (float*)d_out;

    const int M = 672, Dd = 256, FFd = 1024;
    char* ws = (char*)d_ws;
    size_t off = 0;
    auto take = [&](size_t bytes) -> char* {
        char* p = ws + off;
        off = (off + bytes + 255) & ~(size_t)255;
        return p;
    };
    float* X   = (float*)take((size_t)M * Dd * 4);
    float* QK  = (float*)take((size_t)M * Dd * 4);
    float* QKV = (float*)take((size_t)M * 768 * 4);
    float* T1  = (float*)take((size_t)M * FFd * 4);
    float* T2  = (float*)take((size_t)M * FFd * 4);
    float* FUS = (float*)take((size_t)M * Dd * 4);
    float* REF = (float*)take((size_t)M * 3 * 4);
    unsigned short* ABF  = (unsigned short*)take((size_t)M * FFd * 2);
    unsigned short* ABF2 = (unsigned short*)take((size_t)M * Dd * 2);
    unsigned short* WBF  = (unsigned short*)take((size_t)FFd * Dd * 2);
    unsigned short* PW0B = (unsigned short*)take((size_t)Dd * Dd * 2);
    unsigned short* PW1B = (unsigned short*)take((size_t)Dd * Dd * 2);

    (void)hipMemcpyAsync(X, tgt, (size_t)M * Dd * 4, hipMemcpyDeviceToDevice, stream);
    (void)hipMemcpyAsync(REF, refp, (size_t)M * 3 * 4, hipMemcpyDeviceToDevice, stream);

    auto cvt = [&](const float* s, unsigned short* d, int n) {
        k_cvt<<<dim3((n + 255) / 256), dim3(256), 0, stream>>>(s, d, n);
    };
    auto gemm = [&](const unsigned short* A, const unsigned short* Wb, const float* bias,
                    float* C, int Mq, int Nq, int Kq, int ldc, int relu) {
        k_gemm<<<dim3(Nq / 64, Mq / 16), dim3(128), 0, stream>>>(A, Wb, bias, C, Kq, ldc, relu);
    };

    cvt(pW0, PW0B, Dd * Dd);
    cvt(pW1, PW1B, Dd * Dd);

    for (int l = 0; l < 6; ++l) {
        // ---- self-attention ----
        k_add<<<dim3((M * Dd + 255) / 256), dim3(256), 0, stream>>>(X, qpos, QK, M * Dd);
        cvt(QK, ABF, M * Dd);
        cvt(X, ABF2, M * Dd);
        cvt(Wqkv + (size_t)l * 768 * Dd, WBF, 768 * Dd);
        gemm(ABF,  WBF,            bqkv + (size_t)l * 768,       QKV,       M, 512, Dd, 768, 0);
        gemm(ABF2, WBF + 512 * Dd, bqkv + (size_t)l * 768 + 512, QKV + 512, M, 256, Dd, 768, 0);
        k_attn<<<dim3(21, 8, 2), dim3(32), 0, stream>>>(QKV, T1);
        cvt(T1, ABF, M * Dd);
        cvt(Wo + (size_t)l * Dd * Dd, WBF, Dd * Dd);
        gemm(ABF, WBF, bo + (size_t)l * Dd, T2, M, Dd, Dd, Dd, 0);
        k_ln<<<dim3(M), dim3(256), 0, stream>>>(X, T2, ln1g + (size_t)l * Dd, ln1b + (size_t)l * Dd);
        // ---- projective attention ----
        k_proj_sample<<<dim3(336, 2), dim3(256), 0, stream>>>(REF, camR, camT, camK, f0, f1, f2, FUS);
        cvt(FUS, ABF, M * Dd);
        cvt(Wproj + (size_t)l * Dd * Dd, WBF, Dd * Dd);
        gemm(ABF, WBF, bproj + (size_t)l * Dd, T2, M, Dd, Dd, Dd, 0);
        k_ln<<<dim3(M), dim3(256), 0, stream>>>(X, T2, png + (size_t)l * Dd, pnb + (size_t)l * Dd);
        // ---- FFN ----
        cvt(X, ABF, M * Dd);
        cvt(W1 + (size_t)l * FFd * Dd, WBF, FFd * Dd);
        gemm(ABF, WBF, b1 + (size_t)l * FFd, T1, M, FFd, Dd, FFd, 1);
        cvt(T1, ABF, M * FFd);
        cvt(W2 + (size_t)l * Dd * FFd, WBF, Dd * FFd);
        gemm(ABF, WBF, b2 + (size_t)l * Dd, T2, M, Dd, FFd, Dd, 0);
        k_ln<<<dim3(M), dim3(256), 0, stream>>>(X, T2, ln2g + (size_t)l * Dd, ln2b + (size_t)l * Dd);
        // ---- pose + cls heads ----
        cvt(X, ABF, M * Dd);
        gemm(ABF, PW0B, pb0, T1, M, Dd, Dd, Dd, 1);
        cvt(T1, ABF, M * Dd);
        gemm(ABF, PW1B, pb1, T2, M, Dd, Dd, Dd, 1);
        k_pose_head<<<dim3((M * 3 + 255) / 256), dim3(256), 0, stream>>>(T2, pW2, pb2, REF, outp, l);
        k_cls_head<<<dim3((M + 255) / 256), dim3(256), 0, stream>>>(X, clsW, clsb, outp, l);
    }
}
